// TransducerPredictor_62766652063946
// MI455X (gfx1250) — matrix operand order repaired and checked by reading, not compiled, not run on hardware
//
#include <hip/hip_runtime.h>
#include <stdint.h>

// ---------------------------------------------------------------------------
// Transducer predictor (3-layer LSTM) for MI455X / gfx1250.
//   - Input projections (B*T,1024)@(1024,4096) per layer: bf16 WMMA, fp32 acc.
//   - Recurrent per-step GEMM (64,1024)@(1024,4096): bf16 WMMA; R lives in L2.
//   - Per-step fused kernel: LN(4096) + gates + LN(1024) + masked state update.
// Weights pre-transposed to N-major bf16 so every WMMA fragment is two
// contiguous global_load_b128 per lane (same contiguous k order for A and B).
// ---------------------------------------------------------------------------

#define L_LAYERS 3
#define B_BATCH  64
#define U_SEQ    512
#define D_DIM    1024
#define D4       4096
#define LN_EPS   0.001f

typedef __attribute__((ext_vector_type(16))) __bf16 v16bf;
typedef __attribute__((ext_vector_type(8)))  float  v8f;

union Frag { uint4 q[2]; v16bf v; };

static __device__ __forceinline__ unsigned short f2bf(float f) {
    unsigned int u = __float_as_uint(f);
    unsigned int r = (u + 0x7FFFu + ((u >> 16) & 1u)) >> 16;   // round-nearest-even
    return (unsigned short)r;
}
static __device__ __forceinline__ float sigmoidf(float x) {
    return 1.0f / (1.0f + __expf(-x));
}

// ---------------------------------------------------------------------------
// fp32 (L, D, 4D) row-major  ->  bf16 (L, 4D, D) "N-major" transpose
// ---------------------------------------------------------------------------
__global__ void wt_convert_kernel(const float* __restrict__ W,
                                  unsigned short* __restrict__ WT, long total) {
    long id = (long)blockIdx.x * 256 + threadIdx.x;
    if (id >= total) return;
    int  k    = (int)(id % D_DIM);
    long rest = id / D_DIM;
    int  n    = (int)(rest % D4);
    int  l    = (int)(rest / D4);
    const float* Wl = W + (long)l * D_DIM * D4;
    WT[id] = f2bf(Wl[(long)k * D4 + n]);
}

// ---------------------------------------------------------------------------
// embedding gather -> bf16 activations (B*U, D)
// ---------------------------------------------------------------------------
__global__ void gather_kernel(const int* __restrict__ targets,
                              const float* __restrict__ embed,
                              unsigned short* __restrict__ xbf) {
    int row = blockIdx.x;                       // b*U + u
    int tok = targets[row];
    const float* e = embed + (long)tok * D_DIM;
    unsigned short* o = xbf + (long)row * D_DIM;
    for (int d = threadIdx.x; d < D_DIM; d += 256) o[d] = f2bf(e[d]);
}

// ---------------------------------------------------------------------------
// bf16 WMMA GEMM:  C(M,4096) = A(M,1024) * Bt(4096,1024)^T
//   block = 128 thr (4 waves); block tile 64(M) x 64(N); wave tile 16 x 64.
//   A row-major (K=1024), Bt N-major (each row = one output column, K=1024).
//   Contiguous k order used here, same for A and B (16-bit A / B, wave32):
//     A: lane<16 -> M=lane, v0..7=K(k0..k0+15);  lane>=16 -> K(k0+16..k0+31)
//     B: lane<16 -> N=lane, v0..7=K(k0..k0+15);  lane>=16 -> K(k0+16..k0+31)
// ---------------------------------------------------------------------------
__global__ void __launch_bounds__(128)
gemm_bf16_kernel(const unsigned short* __restrict__ A,
                 const unsigned short* __restrict__ Bt,
                 float* __restrict__ C) {
    const int K = D_DIM, N = D4;
    const int wave = threadIdx.x >> 5;
    const int lane = threadIdx.x & 31;
    const int half = lane >> 4;
    const int l16  = lane & 15;
    const int m0 = blockIdx.y * 64 + wave * 16;
    const int n0 = blockIdx.x * 64;

    v8f acc0 = {}, acc1 = {}, acc2 = {}, acc3 = {};

    const unsigned short* arow  = A  + (long)(m0 + l16) * K + half * 16;
    const unsigned short* brow0 = Bt + (long)(n0 +  0 + l16) * K + half * 16;
    const unsigned short* brow1 = Bt + (long)(n0 + 16 + l16) * K + half * 16;
    const unsigned short* brow2 = Bt + (long)(n0 + 32 + l16) * K + half * 16;
    const unsigned short* brow3 = Bt + (long)(n0 + 48 + l16) * K + half * 16;

    for (int k0 = 0; k0 < K; k0 += 32) {
        __builtin_prefetch(arow  + k0 + 128, 0, 1);   // global_prefetch_b8
        __builtin_prefetch(brow0 + k0 + 128, 0, 1);

        Frag a, b0, b1, b2, b3;
        const uint4* ap = (const uint4*)(arow + k0);
        a.q[0] = ap[0]; a.q[1] = ap[1];               // +16B: K block at k0+8
        const uint4* bp0 = (const uint4*)(brow0 + k0);
        b0.q[0] = bp0[0]; b0.q[1] = bp0[1];
        const uint4* bp1 = (const uint4*)(brow1 + k0);
        b1.q[0] = bp1[0]; b1.q[1] = bp1[1];
        const uint4* bp2 = (const uint4*)(brow2 + k0);
        b2.q[0] = bp2[0]; b2.q[1] = bp2[1];
        const uint4* bp3 = (const uint4*)(brow3 + k0);
        b3.q[0] = bp3[0]; b3.q[1] = bp3[1];

        acc0 = __builtin_amdgcn_wmma_f32_16x16x32_bf16(false, a.v, false, b0.v,
                                                       (short)0, acc0, false, false);
        acc1 = __builtin_amdgcn_wmma_f32_16x16x32_bf16(false, a.v, false, b1.v,
                                                       (short)0, acc1, false, false);
        acc2 = __builtin_amdgcn_wmma_f32_16x16x32_bf16(false, a.v, false, b2.v,
                                                       (short)0, acc2, false, false);
        acc3 = __builtin_amdgcn_wmma_f32_16x16x32_bf16(false, a.v, false, b3.v,
                                                       (short)0, acc3, false, false);
    }

    // C/D layout: VGPR r -> row m0 + half*8 + r, col n0 + tile*16 + l16
    float* crow = C + (long)(m0 + half * 8) * N + n0 + l16;
#pragma unroll
    for (int r = 0; r < 8; r++) {
        crow[(long)r * N +  0] = acc0[r];
        crow[(long)r * N + 16] = acc1[r];
        crow[(long)r * N + 32] = acc2[r];
        crow[(long)r * N + 48] = acc3[r];
    }
}

// ---------------------------------------------------------------------------
// Row LayerNorm over width 4096, in place (one block per row).
// ---------------------------------------------------------------------------
__global__ void __launch_bounds__(256)
ln_rows_kernel(float* __restrict__ X, const float* __restrict__ g,
               const float* __restrict__ bta) {
    __shared__ float r1[256], r2[256];
    float* x = X + (long)blockIdx.x * D4;
    const int tid = threadIdx.x;
    float s = 0.f, s2 = 0.f, v[16];
#pragma unroll
    for (int i = 0; i < 16; i++) {
        float t = x[tid + i * 256];
        v[i] = t; s += t; s2 += t * t;
    }
    r1[tid] = s; r2[tid] = s2; __syncthreads();
    for (int off = 128; off > 0; off >>= 1) {
        if (tid < off) { r1[tid] += r1[tid + off]; r2[tid] += r2[tid + off]; }
        __syncthreads();
    }
    float mean = r1[0] * (1.0f / D4);
    float rstd = rsqrtf(r2[0] * (1.0f / D4) - mean * mean + LN_EPS);
#pragma unroll
    for (int i = 0; i < 16; i++) {
        int j = tid + i * 256;
        x[j] = (v[i] - mean) * rstd * g[j] + bta[j];
    }
}

__global__ void init_state_kernel(float* __restrict__ h, unsigned short* __restrict__ hbf,
                                  float* __restrict__ c, float* __restrict__ po) {
    int id = blockIdx.x * 256 + threadIdx.x;     // B*D = 65536 elements
    h[id] = 0.f; c[id] = 0.f; po[id] = 0.f; hbf[id] = 0;
}

// ---------------------------------------------------------------------------
// One LSTM timestep (post-GEMM). One block per batch row b.
//   z = xz[b,t,:] + LN_4096(hR[b,:])*rn_g+rn_b + bias
//   c_new = LN_1024(sig(zf)*c + sig(zi)*tanh(zg)) ; h_new = sig(zo)*tanh(c_new)
//   masked update of h, c, prev_out; writes fp32 output and bf16 next-layer x.
// ---------------------------------------------------------------------------
__global__ void __launch_bounds__(256)
step_kernel(const float* __restrict__ xz, const float* __restrict__ hR,
            const float* __restrict__ bias,
            const float* __restrict__ rng, const float* __restrict__ rnb,
            const float* __restrict__ sng, const float* __restrict__ snb,
            const int* __restrict__ lens,
            float* __restrict__ h, unsigned short* __restrict__ hbf,
            float* __restrict__ c, float* __restrict__ po,
            float* __restrict__ yout, unsigned short* __restrict__ ybf, int t) {
    __shared__ float zsh[D4];
    __shared__ float r1[256], r2[256];
    const int b = blockIdx.x, tid = threadIdx.x;
    const float* hr = hR + (long)b * D4;

    // LN stats over hR row (4096)
    float s = 0.f, s2 = 0.f, hv[16];
#pragma unroll
    for (int i = 0; i < 16; i++) {
        float v = hr[tid + i * 256];
        hv[i] = v; s += v; s2 += v * v;
    }
    r1[tid] = s; r2[tid] = s2; __syncthreads();
    for (int off = 128; off > 0; off >>= 1) {
        if (tid < off) { r1[tid] += r1[tid + off]; r2[tid] += r2[tid + off]; }
        __syncthreads();
    }
    float mean = r1[0] * (1.0f / D4);
    float rstd = rsqrtf(r2[0] * (1.0f / D4) - mean * mean + LN_EPS);

    const float* xzr = xz + ((long)b * U_SEQ + t) * D4;
#pragma unroll
    for (int i = 0; i < 16; i++) {
        int j = tid + i * 256;
        zsh[j] = xzr[j] + (hv[i] - mean) * rstd * rng[j] + rnb[j] + bias[j];
    }
    __syncthreads();

    // gates -> pre-LN cell, stats over 1024
    float ac[4]; float cs = 0.f, cs2 = 0.f;
#pragma unroll
    for (int i = 0; i < 4; i++) {
        int d = tid + i * 256;
        float zi = zsh[d], zf = zsh[D_DIM + d], zg = zsh[2 * D_DIM + d];
        float a = sigmoidf(zf) * c[(long)b * D_DIM + d] + sigmoidf(zi) * tanhf(zg);
        ac[i] = a; cs += a; cs2 += a * a;
    }
    __syncthreads();
    r1[tid] = cs; r2[tid] = cs2; __syncthreads();
    for (int off = 128; off > 0; off >>= 1) {
        if (tid < off) { r1[tid] += r1[tid + off]; r2[tid] += r2[tid + off]; }
        __syncthreads();
    }
    float mean2 = r1[0] * (1.0f / D_DIM);
    float rstd2 = rsqrtf(r2[0] * (1.0f / D_DIM) - mean2 * mean2 + LN_EPS);

    const bool m = t < lens[b];
    const long yrow = ((long)b * U_SEQ + t) * D_DIM;
#pragma unroll
    for (int i = 0; i < 4; i++) {
        int d = tid + i * 256;
        long hd = (long)b * D_DIM + d;
        float cl = (ac[i] - mean2) * rstd2 * sng[d] + snb[d];
        float zo = zsh[3 * D_DIM + d];
        float hn = sigmoidf(zo) * tanhf(cl);
        float h2 = m ? hn : h[hd];
        float c2 = m ? cl : c[hd];
        float o  = m ? hn : po[hd];
        h[hd] = h2; c[hd] = c2; po[hd] = o;
        hbf[hd] = f2bf(h2);
        yout[yrow + d] = o;
        ybf[yrow + d]  = f2bf(o);            // becomes next layer's GEMM input
    }
}

// ---------------------------------------------------------------------------
extern "C" void kernel_launch(void* const* d_in, const int* in_sizes, int n_in,
                              void* d_out, int out_size, void* d_ws, size_t ws_size,
                              hipStream_t stream) {
    (void)in_sizes; (void)n_in; (void)out_size; (void)ws_size;
    const int*   targets = (const int*)  d_in[0];
    const int*   lens    = (const int*)  d_in[1];
    const float* embed   = (const float*)d_in[2];
    const float* W       = (const float*)d_in[3];
    const float* R       = (const float*)d_in[4];
    const float* bias    = (const float*)d_in[5];
    const float* kng     = (const float*)d_in[6];
    const float* knb     = (const float*)d_in[7];
    const float* rng     = (const float*)d_in[8];
    const float* rnb     = (const float*)d_in[9];
    const float* sng     = (const float*)d_in[10];
    const float* snb     = (const float*)d_in[11];
    float* yout = (float*)d_out;

    // workspace carve-out (~660 MB total)
    char* p = (char*)d_ws;
    auto alloc = [&](size_t bytes) -> char* {
        char* r = p; p += (bytes + 255) & ~(size_t)255; return r;
    };
    unsigned short* WT  = (unsigned short*)alloc((size_t)L_LAYERS * D4 * D_DIM * 2);
    unsigned short* RT  = (unsigned short*)alloc((size_t)L_LAYERS * D4 * D_DIM * 2);
    unsigned short* xbf = (unsigned short*)alloc((size_t)B_BATCH * U_SEQ * D_DIM * 2);
    float* XZ = (float*)alloc((size_t)B_BATCH * U_SEQ * D4 * 4);
    float* hR = (float*)alloc((size_t)B_BATCH * D4 * 4);
    float* h  = (float*)alloc((size_t)B_BATCH * D_DIM * 4);
    unsigned short* hbf = (unsigned short*)alloc((size_t)B_BATCH * D_DIM * 2);
    float* c  = (float*)alloc((size_t)B_BATCH * D_DIM * 4);
    float* po = (float*)alloc((size_t)B_BATCH * D_DIM * 4);

    const long totalW = (long)L_LAYERS * D4 * D_DIM;
    wt_convert_kernel<<<(unsigned)((totalW + 255) / 256), 256, 0, stream>>>(W, WT, totalW);
    wt_convert_kernel<<<(unsigned)((totalW + 255) / 256), 256, 0, stream>>>(R, RT, totalW);
    gather_kernel<<<B_BATCH * U_SEQ, 256, 0, stream>>>(targets, embed, xbf);

    for (int l = 0; l < L_LAYERS; l++) {
        const unsigned short* WTl = WT + (long)l * D4 * D_DIM;
        const unsigned short* RTl = RT + (long)l * D4 * D_DIM;

        // XZ = LN(x @ W[l])  -- one big WMMA GEMM + row LN
        dim3 g1(D4 / 64, (B_BATCH * U_SEQ) / 64);
        gemm_bf16_kernel<<<g1, 128, 0, stream>>>(xbf, WTl, XZ);
        ln_rows_kernel<<<B_BATCH * U_SEQ, 256, 0, stream>>>(XZ, kng + l * D4, knb + l * D4);

        init_state_kernel<<<(B_BATCH * D_DIM) / 256, 256, 0, stream>>>(h, hbf, c, po);

        dim3 g2(D4 / 64, 1);                 // M = 64 -> single block row
        for (int t = 0; t < U_SEQ; t++) {
            gemm_bf16_kernel<<<g2, 128, 0, stream>>>(hbf, RTl, hR);
            step_kernel<<<B_BATCH, 256, 0, stream>>>(
                XZ, hR, bias + l * D4,
                rng + l * D4, rnb + l * D4, sng + l * D_DIM, snb + l * D_DIM,
                lens, h, hbf, c, po, yout, xbf, t);
        }
    }
}
